// PGNN_54039278518603
// MI455X (gfx1250) — matrix operand order repaired and checked by reading, not compiled, not run on hardware
//
#include <hip/hip_runtime.h>
#include <hip/hip_bf16.h>

// Problem sizes (match reference)
#define B_SZ  4096
#define DIN   1024
#define DOUT  1024
#define P     16

// GEMM tiling
#define BM 128
#define BN 128
#define BK 32
#define LDS_S 40   // halves per LDS row (32 data + 8 pad), keeps 16B alignment
#define NSTAGE 3   // async pipeline depth

typedef __attribute__((ext_vector_type(16))) __bf16 v16bf;
typedef __attribute__((ext_vector_type(8)))  float  v8f;

union FragCast { uint4 u[2]; v16bf v; };

__device__ __forceinline__ unsigned short f2bf(float f) {
    union { float f; unsigned u; } cv; cv.f = f;
    unsigned u = cv.u;
    u += 0x7FFFu + ((u >> 16) & 1u);   // round-to-nearest-even
    return (unsigned short)(u >> 16);
}
__device__ __forceinline__ unsigned pack2(float lo, float hi) {
    return (unsigned)f2bf(lo) | ((unsigned)f2bf(hi) << 16);
}

// gfx1250 async global->LDS DMA (16B per lane), tracked by ASYNCcnt
__device__ __forceinline__ void async_load_b128(unsigned lds_off, const void* gptr) {
    asm volatile("global_load_async_to_lds_b128 %0, %1, off"
                 :: "v"(lds_off), "v"(gptr) : "memory");
}

// ---------------------------------------------------------------------------
// Kernel 0: f32 -> bf16 bulk conversion (one-time; x and W)
// ---------------------------------------------------------------------------
__global__ __launch_bounds__(256) void k_cvt(const float* __restrict__ src,
                                             unsigned short* __restrict__ dst) {
    const int idx = (blockIdx.x * 256 + threadIdx.x) * 8;
    float4 a = ((const float4*)(src + idx))[0];
    float4 b = ((const float4*)(src + idx))[1];
    uint4 u;
    u.x = pack2(a.x, a.y);
    u.y = pack2(a.z, a.w);
    u.z = pack2(b.x, b.y);
    u.w = pack2(b.z, b.w);
    *(uint4*)(dst + idx) = u;
}

// ---------------------------------------------------------------------------
// Kernel 1: AsumT[d*P + p] = sum_k A[p,k,d]  ;  bsum[p] = sum_k b[p,k]
// ---------------------------------------------------------------------------
__global__ void k_asum(const float* __restrict__ A, const float* __restrict__ b,
                       float* __restrict__ AsumT, float* __restrict__ bsum) {
    const int p = blockIdx.x;
    const float* Ap = A + (size_t)p * DIN * DIN;
    for (int d = threadIdx.x; d < DIN; d += blockDim.x) {
        float s = 0.f;
        for (int k = 0; k < DIN; ++k) s += Ap[(size_t)k * DIN + d];
        AsumT[d * P + p] = s;
    }
    if (threadIdx.x == 0) {
        float s = 0.f;
        const float* bp = b + p * DIN;
        for (int k = 0; k < DIN; ++k) s += bp[k];
        bsum[p] = s;
    }
}

// ---------------------------------------------------------------------------
// Kernel 2: gate m[b,p] (row-major) and mT[p,b] (transposed copy for blending)
// ---------------------------------------------------------------------------
__global__ __launch_bounds__(256) void k_gate(
    const float* __restrict__ x, const float* __restrict__ AsumT,
    const float* __restrict__ bsum, float* __restrict__ m,
    float* __restrict__ mT) {
    const int lane = threadIdx.x & 31;
    const int wave = threadIdx.x >> 5;
    const int row  = blockIdx.x * 8 + wave;
    const float* xr = x + (size_t)row * DIN;

    float acc[P];
#pragma unroll
    for (int p = 0; p < P; ++p) acc[p] = 0.f;

    for (int d = lane; d < DIN; d += 32) {
        float xv = xr[d];
        const float4* at = (const float4*)(AsumT + d * P);
#pragma unroll
        for (int q = 0; q < 4; ++q) {
            float4 a = at[q];
            acc[q*4+0] += xv * a.x;
            acc[q*4+1] += xv * a.y;
            acc[q*4+2] += xv * a.z;
            acc[q*4+3] += xv * a.w;
        }
    }
#pragma unroll
    for (int p = 0; p < P; ++p) {
        float v = acc[p];
#pragma unroll
        for (int off = 16; off > 0; off >>= 1) v += __shfl_down(v, off, 32);
        acc[p] = v;
    }
    if (lane == 0) {
        float s = 0.f, g[P];
#pragma unroll
        for (int p = 0; p < P; ++p) {
            float sc = acc[p] - bsum[p];
            float gg = 1.f / (1.f + __expf(sc));   // sigmoid(-sc)
            g[p] = gg; s += gg;
        }
        float inv = 1.f / (s + 1e-6f);
#pragma unroll
        for (int p = 0; p < P; ++p) {
            float gv = g[p] * inv;
            m[row * P + p] = gv;
            mT[(size_t)p * B_SZ + row] = gv;
        }
    }
}

// ---------------------------------------------------------------------------
// WMMA pass over one staged LDS tile pair (K=32): 4 M-tiles x 2 N-tiles/wave.
// A frag (16x32 bf16): lane<16 -> row=lane, K{0..7,16..23}; lane>=16 -> K{8..15,24..31}
// B frag (32x16 bf16): lane<16 -> col=lane, K{0..7,16..23}; lane>=16 -> K{8..15,24..31}
// ---------------------------------------------------------------------------
__device__ __forceinline__ void mma_pass(const unsigned short* sX,
                                         const unsigned short* sW,
                                         int wr, int wc, int lane,
                                         v8f acc[4][2]) {
    const int l15 = lane & 15;
    const int alo = (lane < 16) ? 0 : 8;
    const int blo = (lane < 16) ? 0 : 8;

    v16bf afrag[4];
#pragma unroll
    for (int i = 0; i < 4; ++i) {
        const unsigned short* base = sX + (wr * 64 + i * 16 + l15) * LDS_S;
        FragCast t;
        t.u[0] = *(const uint4*)(base + alo);
        t.u[1] = *(const uint4*)(base + alo + 16);
        afrag[i] = t.v;
    }
    v16bf bfrag[2];
#pragma unroll
    for (int j = 0; j < 2; ++j) {
        const unsigned short* base = sW + ((wc * 2 + j) * 16 + l15) * LDS_S;
        FragCast t;
        t.u[0] = *(const uint4*)(base + blo);
        t.u[1] = *(const uint4*)(base + blo + 16);
        bfrag[j] = t.v;
    }
#pragma unroll
    for (int i = 0; i < 4; ++i)
#pragma unroll
        for (int j = 0; j < 2; ++j)
            acc[i][j] = __builtin_amdgcn_wmma_f32_16x16x32_bf16(
                false, afrag[i], false, bfrag[j],
                (short)0, acc[i][j], false, false);
}

// ---------------------------------------------------------------------------
// Kernel 3: out = sum_p m[:,p] * (x @ W[p]^T) + m @ c
// Fused-K bf16 WMMA GEMM, 3-deep async-LDS pipeline, f32 gate blend.
// ---------------------------------------------------------------------------
__global__ __launch_bounds__(256) void k_moe(
    const unsigned short* __restrict__ xbf, const unsigned short* __restrict__ Wbf,
    const float* __restrict__ c, const float* __restrict__ m,
    const float* __restrict__ mT, float* __restrict__ out) {

    __shared__ __align__(16) unsigned short sX[NSTAGE][BM * LDS_S];   // 3 x 10 KB
    __shared__ __align__(16) unsigned short sW[NSTAGE][BN * LDS_S];   // 3 x 10 KB

    const int tid  = threadIdx.x;
    const int lane = tid & 31;
    const int wave = tid >> 5;
    const int wr   = wave >> 2;   // 0..1 : 64-row strip
    const int wc   = wave & 3;    // 0..3 : 32-col strip
    const int bm   = blockIdx.x * BM;
    const int bn   = blockIdx.y * BN;
    const int l15  = lane & 15;
    const int rofs = (lane < 16) ? 0 : 8;   // C/D layout row offset

    v8f acc[4][2], accp[4][2];
#pragma unroll
    for (int i = 0; i < 4; ++i)
#pragma unroll
        for (int j = 0; j < 2; ++j) {
            acc[i][j]  = (v8f){0.f,0.f,0.f,0.f,0.f,0.f,0.f,0.f};
            accp[i][j] = (v8f){0.f,0.f,0.f,0.f,0.f,0.f,0.f,0.f};
        }

    // issue one stage: 4 async b128 DMAs per thread (x tile + W tile)
    auto issue = [&](int pk, int buf) {
        const int p  = pk >> 5;           // DIN/BK == 32 chunks per expert
        const int d0 = (pk & 31) * BK;
#pragma unroll
        for (int i = 0; i < 2; ++i) {
            const int seg = tid + i * 256;     // 0..511
            const int row = seg >> 2;          // 0..127
            const int cs  = (seg & 3) * 8;     // 0,8,16,24 halves

            const unsigned short* gx = xbf + (size_t)(bm + row) * DIN + d0 + cs;
            const unsigned short* gw = Wbf + ((size_t)p * DOUT + bn + row) * DIN + d0 + cs;
            async_load_b128((unsigned)(size_t)&sX[buf][row * LDS_S + cs], gx);
            async_load_b128((unsigned)(size_t)&sW[buf][row * LDS_S + cs], gw);
        }
    };

    const int NCHUNK = P * (DIN / BK);   // 512 fused-K chunks
    issue(0, 0);
    issue(1, 1);
    int buf = 0;
    for (int pk = 0; pk < NCHUNK; ++pk) {
        if (pk + 2 < NCHUNK) {
            int ibuf = buf + 2; if (ibuf >= NSTAGE) ibuf -= NSTAGE;
            issue(pk + 2, ibuf);
            asm volatile("s_wait_asynccnt 0x8" ::: "memory");  // stage pk done
        } else if (pk + 1 < NCHUNK) {
            asm volatile("s_wait_asynccnt 0x4" ::: "memory");
        } else {
            asm volatile("s_wait_asynccnt 0x0" ::: "memory");
        }
        __syncthreads();

        mma_pass(sX[buf], sW[buf], wr, wc, lane, accp);

        // expert boundary: blend accp into acc with f32 gate weights (from mT)
        if ((pk & 31) == 31) {
            const int p = pk >> 5;
            const float* mTp = mT + (size_t)p * B_SZ + bm + wr * 64 + rofs;
#pragma unroll
            for (int i = 0; i < 4; ++i) {
                float4 m0 = ((const float4*)(mTp + i * 16))[0];
                float4 m1 = ((const float4*)(mTp + i * 16))[1];
                float mrow[8] = {m0.x, m0.y, m0.z, m0.w, m1.x, m1.y, m1.z, m1.w};
#pragma unroll
                for (int j = 0; j < 2; ++j) {
#pragma unroll
                    for (int v = 0; v < 8; ++v)
                        acc[i][j][v] += mrow[v] * accp[i][j][v];
                    accp[i][j] = (v8f){0.f,0.f,0.f,0.f,0.f,0.f,0.f,0.f};
                }
            }
        }
        __syncthreads();
        ++buf; if (buf >= NSTAGE) buf = 0;
    }

    // ---- bias: one extra K=16 chunk, A = m[b,0..15], B^T = c[p,o], into acc ----
    {
        const int row = tid >> 1;        // 0..127
        const int hb  = tid & 1;
        if (hb == 0) {
            const float* mr = m + (size_t)(bm + row) * P;
            uint4 u0, u1;
            u0.x = pack2(mr[0],  mr[1]);  u0.y = pack2(mr[2],  mr[3]);
            u0.z = pack2(mr[4],  mr[5]);  u0.w = pack2(mr[6],  mr[7]);
            u1.x = pack2(mr[8],  mr[9]);  u1.y = pack2(mr[10], mr[11]);
            u1.z = pack2(mr[12], mr[13]); u1.w = pack2(mr[14], mr[15]);
            *(uint4*)(&sX[0][row * LDS_S])     = u0;
            *(uint4*)(&sX[0][row * LDS_S + 8]) = u1;

            unsigned short h[P];
#pragma unroll
            for (int q = 0; q < P; ++q) h[q] = f2bf(c[q * DOUT + bn + row]);
            uint4 w0, w1;
            w0.x = (unsigned)h[0]  | ((unsigned)h[1]  << 16);
            w0.y = (unsigned)h[2]  | ((unsigned)h[3]  << 16);
            w0.z = (unsigned)h[4]  | ((unsigned)h[5]  << 16);
            w0.w = (unsigned)h[6]  | ((unsigned)h[7]  << 16);
            w1.x = (unsigned)h[8]  | ((unsigned)h[9]  << 16);
            w1.y = (unsigned)h[10] | ((unsigned)h[11] << 16);
            w1.z = (unsigned)h[12] | ((unsigned)h[13] << 16);
            w1.w = (unsigned)h[14] | ((unsigned)h[15] << 16);
            *(uint4*)(&sW[0][row * LDS_S])     = w0;
            *(uint4*)(&sW[0][row * LDS_S + 8]) = w1;
        } else {
            uint4 z = {0u, 0u, 0u, 0u};
            *(uint4*)(&sX[0][row * LDS_S + 16]) = z;
            *(uint4*)(&sX[0][row * LDS_S + 24]) = z;
            *(uint4*)(&sW[0][row * LDS_S + 16]) = z;
            *(uint4*)(&sW[0][row * LDS_S + 24]) = z;
        }
        __syncthreads();
        mma_pass(sX[0], sW[0], wr, wc, lane, acc);
    }

    // ---- store: row = tile*16 + v + rofs, col = tileN*16 + (lane&15) ----
#pragma unroll
    for (int i = 0; i < 4; ++i) {
        const int rbase = bm + wr * 64 + i * 16 + rofs;
#pragma unroll
        for (int j = 0; j < 2; ++j) {
            const int col = bn + (wc * 2 + j) * 16 + l15;
#pragma unroll
            for (int v = 0; v < 8; ++v)
                out[(size_t)(rbase + v) * DOUT + col] = acc[i][j][v];
        }
    }
}

// ---------------------------------------------------------------------------
extern "C" void kernel_launch(void* const* d_in, const int* in_sizes, int n_in,
                              void* d_out, int out_size, void* d_ws, size_t ws_size,
                              hipStream_t stream) {
    const float* x = (const float*)d_in[0];
    const float* A = (const float*)d_in[1];
    const float* b = (const float*)d_in[2];
    const float* W = (const float*)d_in[3];
    const float* c = (const float*)d_in[4];
    float* out = (float*)d_out;

    // workspace layout (floats / halves), all 16B-aligned
    float* AsumT = (float*)d_ws;                         // P*DIN floats
    float* bsum  = AsumT + P * DIN;                      // P floats (+pad)
    float* m     = bsum + P + 12;                        // B_SZ*P floats
    float* mT    = m + (size_t)B_SZ * P;                 // P*B_SZ floats
    unsigned short* xbf = (unsigned short*)(mT + (size_t)P * B_SZ); // B_SZ*DIN bf16
    unsigned short* Wbf = xbf + (size_t)B_SZ * DIN;      // P*DOUT*DIN bf16

    k_cvt<<<(B_SZ * DIN) / 2048, 256, 0, stream>>>(x, xbf);
    k_cvt<<<(P * DOUT * DIN) / 2048, 256, 0, stream>>>(W, Wbf);
    k_asum<<<P, 256, 0, stream>>>(A, b, AsumT, bsum);
    k_gate<<<B_SZ / 8, 256, 0, stream>>>(x, AsumT, bsum, m, mT);
    dim3 grid(B_SZ / BM, DOUT / BN);
    k_moe<<<grid, 256, 0, stream>>>(xbf, Wbf, c, m, mT, out);
}
